// Head_47983374630998
// MI455X (gfx1250) — matrix operand order repaired and checked by reading, not compiled, not run on hardware
//
#include <hip/hip_runtime.h>
#include <hip/hip_bf16.h>

typedef __attribute__((ext_vector_type(16))) __bf16 v16bf;
typedef __attribute__((ext_vector_type(8)))  float  v8f;

#define Bb 8
#define Tt 2048
#define Cc 1024
#define Hh 128

// ---- fragment load helpers -------------------------------------------------
__device__ __forceinline__ v16bf ld2x16(const void* p0, const void* p1) {
  union { v16bf v; uint4 u[2]; } t;
  t.u[0] = *(const uint4*)p0;
  t.u[1] = *(const uint4*)p1;
  return t.v;
}
__device__ __forceinline__ v16bf ld32(const void* p) {
  return *(const v16bf*)p;   // 32B contiguous, 32B aligned
}
__device__ __forceinline__ v8f wmma_bf16(v16bf a, v16bf b, v8f c) {
  return __builtin_amdgcn_wmma_f32_16x16x32_bf16(false, a, false, b, (short)0, c,
                                                 false, false);
}

// ---- weight transpose + fp32->bf16 : WT[n][k] = W[k][n] --------------------
__global__ void wtrans_kernel(const float* __restrict__ W, __bf16* __restrict__ WT) {
  int tid = blockIdx.x * blockDim.x + threadIdx.x;   // 0 .. 128*1024-1
  int n = tid >> 10;          // 0..127
  int k = tid & 1023;         // 0..1023
  WT[(size_t)n * Cc + k] = (__bf16)W[(size_t)k * Hh + n];
}

// ---- projection GEMM: out = x[16384,1024] @ W  (via WT), bf16 out ----------
// transposed==0 : out[row][h] row-major [B*T, H]
// transposed==1 : out[b][h][t]          [B, H, T]
__global__ void proj_kernel(const float* __restrict__ x,
                            const __bf16* __restrict__ WT,
                            __bf16* __restrict__ out,
                            float scale, int transposed) {
  const int lane = threadIdx.x & 31;
  const int wave = threadIdx.x >> 5;
  const int hi   = lane >> 4;       // 0/1 half
  const int ln   = lane & 15;
  const int rowBase = blockIdx.x * 128 + wave * 16;

  v8f acc[8] = {};
  const float* xrow = x + (size_t)(rowBase + ln) * Cc;

  for (int ks = 0; ks < Cc / 32; ++ks) {
    // A fragment: x row (lane&15), K = ks*32 + {0..7,16..23} (+8 for hi half)
    const int c0 = ks * 32 + hi * 8;
    float4 f0 = *(const float4*)(xrow + c0);
    float4 f1 = *(const float4*)(xrow + c0 + 4);
    float4 f2 = *(const float4*)(xrow + c0 + 16);
    float4 f3 = *(const float4*)(xrow + c0 + 20);
    v16bf a;
    a[0]=(__bf16)f0.x; a[1]=(__bf16)f0.y; a[2]=(__bf16)f0.z; a[3]=(__bf16)f0.w;
    a[4]=(__bf16)f1.x; a[5]=(__bf16)f1.y; a[6]=(__bf16)f1.z; a[7]=(__bf16)f1.w;
    a[8]=(__bf16)f2.x; a[9]=(__bf16)f2.y; a[10]=(__bf16)f2.z; a[11]=(__bf16)f2.w;
    a[12]=(__bf16)f3.x; a[13]=(__bf16)f3.y; a[14]=(__bf16)f3.z; a[15]=(__bf16)f3.w;
#pragma unroll
    for (int nt = 0; nt < 8; ++nt) {
      const __bf16* bp = WT + (size_t)(nt * 16 + ln) * Cc + ks * 32 + hi * 8;
      acc[nt] = wmma_bf16(a, ld2x16(bp, bp + 16), acc[nt]);
    }
  }

#pragma unroll
  for (int nt = 0; nt < 8; ++nt)
#pragma unroll
    for (int r = 0; r < 8; ++r) {
      int row = rowBase + r + hi * 8;
      int col = nt * 16 + ln;
      __bf16 v = (__bf16)(acc[nt][r] * scale);
      if (!transposed) out[(size_t)row * Hh + col] = v;
      else {
        int b = row >> 11, t = row & 2047;
        out[((size_t)b * Hh + col) * Tt + t] = v;
      }
    }
}

// ---- flash attention: out = softmax(mask(q k^T)) v -------------------------
// qb [B*T,H] bf16 (pre-scaled by sqrt(C)=32), kb [B*T,H] bf16, vT [B,H,T] bf16
__global__ void attn_kernel(const __bf16* __restrict__ qb,
                            const __bf16* __restrict__ kb,
                            const __bf16* __restrict__ vT,
                            float* __restrict__ out) {
  __shared__ __align__(16) __bf16 Plds[8 * 16 * 128];   // 4KB per wave, private
  const int lane = threadIdx.x & 31;
  const int wave = threadIdx.x >> 5;
  const int hi   = lane >> 4;
  const int ln   = lane & 15;
  const int b    = blockIdx.x >> 4;
  const int it   = blockIdx.x & 15;          // query tile of 128 rows
  const int qrow0 = it * 128 + wave * 16;    // this wave's rows (within batch)
  const size_t base = (size_t)b * Tt;

  // resident Q fragments over H=128 (4 K-steps of 32)
  v16bf qf[4];
  {
    const __bf16* qrow = qb + (base + qrow0 + ln) * Hh;
#pragma unroll
    for (int ks = 0; ks < 4; ++ks) {
      const int c0 = ks * 32 + hi * 16;
      qf[ks] = ld2x16(qrow + c0, qrow + c0 + 8);
    }
  }

  v8f Oacc[8] = {};
  float m[8], l[8];
#pragma unroll
  for (int r = 0; r < 8; ++r) { m[r] = -3.0e38f; l[r] = 0.0f; }

  __bf16* P = Plds + wave * 16 * 128;
  const __bf16* vbase = vT + (size_t)b * Hh * Tt;

  for (int j = 0; j <= it; ++j) {
    const __bf16* kt = kb + (base + (size_t)j * 128) * Hh;
    if (j < it) {  // prefetch next K/V tiles
      __builtin_prefetch(kb + (base + (size_t)(j + 1) * 128) * Hh, 0, 1);
      __builtin_prefetch(vbase + (size_t)ln * Tt + (j + 1) * 128, 0, 1);
    }
    // ---- S = Q @ K^T (keys contiguous per row of kb -> direct B frags) ----
    v8f S[8] = {};
#pragma unroll
    for (int ks = 0; ks < 4; ++ks)
#pragma unroll
      for (int nt = 0; nt < 8; ++nt) {
        const __bf16* kp = kt + (size_t)(nt * 16 + ln) * Hh + ks * 32 + hi * 16;
        S[nt] = wmma_bf16(qf[ks], ld32(kp), S[nt]);
      }
    // ---- causal mask on diagonal tile ----
    if (j == it) {
#pragma unroll
      for (int nt = 0; nt < 8; ++nt) {
        int key = nt * 16 + ln;
#pragma unroll
        for (int r = 0; r < 8; ++r)
          if (key > wave * 16 + r + hi * 8) S[nt][r] = -3.0e38f;
      }
    }
    // ---- online softmax: row max, rescale, P = exp(S-m), row sum ----
#pragma unroll
    for (int r = 0; r < 8; ++r) {
      float v = S[0][r];
#pragma unroll
      for (int nt = 1; nt < 8; ++nt) v = fmaxf(v, S[nt][r]);
      v = fmaxf(v, __shfl_xor(v, 1, 32));
      v = fmaxf(v, __shfl_xor(v, 2, 32));
      v = fmaxf(v, __shfl_xor(v, 4, 32));
      v = fmaxf(v, __shfl_xor(v, 8, 32));      // row max (within 16-lane half)
      float mn = fmaxf(m[r], v);
      float da = m[r] - mn;
      float alpha = (da > -80.0f) ? __expf(da) : 0.0f;
      m[r] = mn;
      l[r] *= alpha;
#pragma unroll
      for (int nt = 0; nt < 8; ++nt) Oacc[nt][r] *= alpha;
      float rs = 0.0f;
#pragma unroll
      for (int nt = 0; nt < 8; ++nt) {
        float arg = S[nt][r] - mn;
        float p = (arg > -80.0f) ? __expf(arg) : 0.0f;
        rs += p;
        P[(r + hi * 8) * 128 + nt * 16 + ln] = (__bf16)p;  // C/D -> LDS
      }
      rs += __shfl_xor(rs, 1, 32);
      rs += __shfl_xor(rs, 2, 32);
      rs += __shfl_xor(rs, 4, 32);
      rs += __shfl_xor(rs, 8, 32);
      l[r] += rs;
    }
    // ---- O += P @ V  (A frags from LDS, B frags contiguous from vT) ----
    const __bf16* vt = vbase + (size_t)j * 128;
#pragma unroll
    for (int ks = 0; ks < 4; ++ks) {
      const __bf16* pr = P + ln * 128 + ks * 32 + hi * 16;
      v16bf af = ld2x16(pr, pr + 8);
#pragma unroll
      for (int nh = 0; nh < 8; ++nh) {
        const __bf16* vp = vt + (size_t)(nh * 16 + ln) * Tt + ks * 32 + hi * 16;
        Oacc[nh] = wmma_bf16(af, ld32(vp), Oacc[nh]);
      }
    }
  }

  // ---- normalize and store fp32 output ----
#pragma unroll
  for (int r = 0; r < 8; ++r) {
    float rl = 1.0f / l[r];
    int row = qrow0 + r + hi * 8;
#pragma unroll
    for (int nt = 0; nt < 8; ++nt)
      out[(base + row) * Hh + nt * 16 + ln] = Oacc[nt][r] * rl;
  }
}

// ---- host launch -----------------------------------------------------------
extern "C" void kernel_launch(void* const* d_in, const int* in_sizes, int n_in,
                              void* d_out, int out_size, void* d_ws, size_t ws_size,
                              hipStream_t stream) {
  (void)in_sizes; (void)n_in; (void)out_size; (void)ws_size;
  const float* x  = (const float*)d_in[0];
  const float* Wk = (const float*)d_in[1];
  const float* Wq = (const float*)d_in[2];
  const float* Wv = (const float*)d_in[3];

  __bf16* WkT = (__bf16*)d_ws;                 // [128,1024]
  __bf16* WqT = WkT + Hh * Cc;
  __bf16* WvT = WqT + Hh * Cc;
  __bf16* qb  = WvT + Hh * Cc;                 // [B*T, H]
  __bf16* kb  = qb + (size_t)Bb * Tt * Hh;
  __bf16* vT  = kb + (size_t)Bb * Tt * Hh;     // [B, H, T]

  wtrans_kernel<<<512, 256, 0, stream>>>(Wk, WkT);
  wtrans_kernel<<<512, 256, 0, stream>>>(Wq, WqT);
  wtrans_kernel<<<512, 256, 0, stream>>>(Wv, WvT);

  proj_kernel<<<Bb * Tt / 128, 256, 0, stream>>>(x, WqT, qb, 32.0f, 0); // sqrt(C)
  proj_kernel<<<Bb * Tt / 128, 256, 0, stream>>>(x, WkT, kb, 1.0f, 0);
  proj_kernel<<<Bb * Tt / 128, 256, 0, stream>>>(x, WvT, vT, 1.0f, 1);

  attn_kernel<<<Bb * (Tt / 128), 256, 0, stream>>>(qb, kb, vT, (float*)d_out);
}
